// CustomLSTM_21500606284409
// MI455X (gfx1250) — matrix operand order repaired and checked by reading, not compiled, not run on hardware
//
#include <hip/hip_runtime.h>
#include <hip/hip_bf16.h>

// ---------------------------------------------------------------------------
// 2-layer LSTM LM forward for MI455X (gfx1250).
// f16 WMMA GEMMs (f32 accum), LDS-tiled big GEMMs, fused recurrent step.
// B=16, T=256, V=32000, E=512, H=1024.
// ---------------------------------------------------------------------------

typedef _Float16 h8   __attribute__((ext_vector_type(8)));
typedef _Float16 v16h __attribute__((ext_vector_type(16)));
typedef float    v8f  __attribute__((ext_vector_type(8)));

#define LSTM_B 16
#define LSTM_T 256
#define LSTM_V 32000
#define LSTM_E 512
#define LSTM_H 1024
#define LSTM_G (4 * LSTM_H)

// LDS-tiled GEMM block shape
#define BM 128
#define BN 256
#define BK 32
#define LDSP 40   // padded row length (halves): 80B stride -> conflict-free, 16B aligned

__device__ __forceinline__ v16h cat8(h8 lo, h8 hi) {
  return __builtin_shufflevector(lo, hi, 0, 1, 2, 3, 4, 5, 6, 7,
                                 8, 9, 10, 11, 12, 13, 14, 15);
}

__device__ __forceinline__ float sigmoidf_(float x) {
  return 1.0f / (1.0f + __expf(-x));
}

// ---------------------------------------------------------------------------
// elementwise prep kernels
// ---------------------------------------------------------------------------
__global__ void k_f32_to_f16(const float* __restrict__ src,
                             _Float16* __restrict__ dst, int n) {
  int i = blockIdx.x * blockDim.x + threadIdx.x;
  if (i < n) dst[i] = (_Float16)src[i];
}

__global__ void k_bias_sum(const float* __restrict__ a,
                           const float* __restrict__ b,
                           float* __restrict__ out, int n) {
  int i = blockIdx.x * blockDim.x + threadIdx.x;
  if (i < n) out[i] = a[i] + b[i];
}

// Embedding gather, stored (T, B, E) as f16 so timestep rows are contiguous.
__global__ void k_embed(const int* __restrict__ tokens,
                        const float* __restrict__ emb,
                        _Float16* __restrict__ xe) {
  int i = blockIdx.x * blockDim.x + threadIdx.x;  // T*B*E
  if (i >= LSTM_T * LSTM_B * LSTM_E) return;
  int t = i / (LSTM_B * LSTM_E);
  int r = i % (LSTM_B * LSTM_E);
  int b = r / LSTM_E;
  int e = r % LSTM_E;
  int tok = tokens[b * LSTM_T + t];  // tokens are (B,T)
  xe[i] = (_Float16)emb[(long)tok * LSTM_E + e];
}

// zero c (f32) and both h ping-pong buffers (f16)
__global__ void k_zero_state(float* __restrict__ c, _Float16* __restrict__ ha,
                             _Float16* __restrict__ hb) {
  int i = blockIdx.x * blockDim.x + threadIdx.x;
  if (i < LSTM_B * LSTM_H) {
    c[i] = 0.0f;
    ha[i] = (_Float16)0.0f;
    hb[i] = (_Float16)0.0f;
  }
}

// ---------------------------------------------------------------------------
// LDS-tiled WMMA GEMM (NT): D[M,N] = A_f16[M,K] * W_f16[N,K]^T + bias
//  block = 256 threads (8 waves, 2x4), block tile BM x BN, K-step BK,
//  double-buffered LDS (60KB), wave tile 64x64 (4x4 accumulators).
//  All 8 fragments (4 A + 4 B) are loaded before the 16-WMMA burst so the
//  matrix pipe issues back-to-back with only graduated s_wait_dscnt at entry.
//  M % BM == 0, N % BN == 0, K % BK == 0 (true for all call sites).
//  remapB > 0: dest row = (m % remapB)*remapT + m/remapB  ((T,B)->(B,T)).
// ---------------------------------------------------------------------------
__global__ __launch_bounds__(256) void wmma_gemm_tiled(
    const _Float16* __restrict__ A, int lda,
    const _Float16* __restrict__ W, int ldw,
    const float* __restrict__ bias,
    float* __restrict__ D, int ldd,
    int K, int remapB, int remapT) {
  __shared__ __align__(16) _Float16 As[2][BM][LDSP];
  __shared__ __align__(16) _Float16 Bs[2][BN][LDSP];

  const int tid  = threadIdx.x;
  const int lane = tid & 31;
  const int wave = tid >> 5;
  const int col  = lane & 15;
  const int hi   = lane >> 4;
  const int waveM = wave >> 2;   // 0..1
  const int waveN = wave & 3;    // 0..3
  const int mBlk = blockIdx.y * BM;
  const int nBlk = blockIdx.x * BN;

  v8f acc[4][4];
#pragma unroll
  for (int mi = 0; mi < 4; ++mi)
#pragma unroll
    for (int ni = 0; ni < 4; ++ni)
#pragma unroll
      for (int r = 0; r < 8; ++r) acc[mi][ni][r] = 0.0f;

  h8 aReg[2], bReg[4];
  // global -> regs staging (each h8 chunk = 8 halves; row = chunk/4, col = chunk%4)
  auto loadG = [&](int k0) {
#pragma unroll
    for (int i = 0; i < 2; ++i) {
      int ch = tid + i * 256;
      aReg[i] = *(const h8*)(A + (long)(mBlk + (ch >> 2)) * lda + k0 + (ch & 3) * 8);
    }
#pragma unroll
    for (int i = 0; i < 4; ++i) {
      int ch = tid + i * 256;
      bReg[i] = *(const h8*)(W + (long)(nBlk + (ch >> 2)) * ldw + k0 + (ch & 3) * 8);
    }
  };
  auto storeL = [&](int buf) {
#pragma unroll
    for (int i = 0; i < 2; ++i) {
      int ch = tid + i * 256;
      *(h8*)&As[buf][ch >> 2][(ch & 3) * 8] = aReg[i];
    }
#pragma unroll
    for (int i = 0; i < 4; ++i) {
      int ch = tid + i * 256;
      *(h8*)&Bs[buf][ch >> 2][(ch & 3) * 8] = bReg[i];
    }
  };

  loadG(0);
  storeL(0);
  __syncthreads();

  int buf = 0;
  for (int k0 = 0; k0 < K; k0 += BK) {
    const int kn = k0 + BK;
    if (kn < K) loadG(kn);  // register prefetch of next K-tile

    // L2 prefetch two K-tiles ahead (global_prefetch_b8 on gfx1250)
    if (k0 + 2 * BK < K) {
      __builtin_prefetch(
          W + (long)(nBlk + (tid >> 2)) * ldw + (k0 + 2 * BK) + (tid & 3) * 8,
          0, 1);
      __builtin_prefetch(
          A + (long)(mBlk + (tid >> 2)) * lda + (k0 + 2 * BK) + (tid & 3) * 8,
          0, 1);
    }

    // all 8 fragments up front, then a 16-WMMA burst
    v16h af[4], bf[4];
#pragma unroll
    for (int mi = 0; mi < 4; ++mi) {
      int row = waveM * 64 + mi * 16 + col;
      h8 lo = *(const h8*)&As[buf][row][hi * 8];
      h8 hh = *(const h8*)&As[buf][row][16 + hi * 8];
      af[mi] = cat8(lo, hh);
    }
#pragma unroll
    for (int ni = 0; ni < 4; ++ni) {
      int row = waveN * 64 + ni * 16 + col;
      h8 lo = *(const h8*)&Bs[buf][row][hi * 8];
      h8 hh = *(const h8*)&Bs[buf][row][16 + hi * 8];
      bf[ni] = cat8(lo, hh);
    }
#pragma unroll
    for (int ni = 0; ni < 4; ++ni)
#pragma unroll
      for (int mi = 0; mi < 4; ++mi)
        acc[mi][ni] = __builtin_amdgcn_wmma_f32_16x16x32_f16(
            false, af[mi], false, bf[ni], (short)0, acc[mi][ni], false, false);

    if (kn < K) {
      storeL(buf ^ 1);  // prev reads of buf^1 were fenced by last barrier
      __syncthreads();
      buf ^= 1;
    }
  }

#pragma unroll
  for (int mi = 0; mi < 4; ++mi)
#pragma unroll
    for (int ni = 0; ni < 4; ++ni)
#pragma unroll
      for (int r = 0; r < 8; ++r) {
        int mm = mBlk + waveM * 64 + mi * 16 + r + hi * 8;
        int nn = nBlk + waveN * 64 + ni * 16 + col;
        float v = acc[mi][ni][r];
        if (bias) v += bias[nn];
        long drow = remapB ? ((long)(mm % remapB) * remapT + (mm / remapB))
                           : (long)mm;
        D[drow * (long)ldd + nn] = v;
      }
}

// ---------------------------------------------------------------------------
// Fused recurrent step: for timestep t,
//   gates(16 x 4H) = hprev @ W_hh^T + gi[t]   (WMMA, f32 accum)
//   then LSTM cell applied entirely in-register.
// Each wave owns a 16-wide j-slice; its 4 accumulators are gates i,f,g,o for
// the SAME (b, j) coordinates (gate g lives at column g*H + j of W_hh / gi).
// grid = 8 blocks x 256 threads = 64 waves = H/16 j-tiles, no divergence.
// hprev/hnext ping-pong across timesteps (block reads all of h, writes slice).
// ---------------------------------------------------------------------------
__global__ __launch_bounds__(256) void k_lstm_step(
    const _Float16* __restrict__ hprev,  // (B,H) f16
    const _Float16* __restrict__ Whh,    // (4H,H) f16
    const float* __restrict__ gi_t,      // (B,4H) f32  (bias folded in)
    float* __restrict__ c,               // (B,H) f32
    _Float16* __restrict__ hnext,        // (B,H) f16
    _Float16* __restrict__ hs,           // (T,B,H) f16
    int t) {
  const int lane = threadIdx.x & 31;
  const int wave = threadIdx.x >> 5;
  const int col  = lane & 15;
  const int hi   = lane >> 4;
  const int j0   = (blockIdx.x * 8 + wave) * 16;  // j-tile origin, < H

  // seed accumulators from gi[t]; acc[s] is gate s (i,f,g,o)
  v8f acc[4];
#pragma unroll
  for (int s = 0; s < 4; ++s)
#pragma unroll
    for (int r = 0; r < 8; ++r)
      acc[s][r] = gi_t[(long)(r + hi * 8) * LSTM_G + s * LSTM_H + j0 + col];

  const _Float16* aRow = hprev + (long)col * LSTM_H;  // A row = batch index
  for (int k0 = 0; k0 < LSTM_H; k0 += 32) {
    h8 alo = *(const h8*)(aRow + k0 + hi * 16);
    h8 ahi = *(const h8*)(aRow + k0 + hi * 16 + 8);
    v16h a = cat8(alo, ahi);
#pragma unroll
    for (int s = 0; s < 4; ++s) {
      const _Float16* wRow =
          Whh + (long)(s * LSTM_H + j0 + col) * LSTM_H + k0 + hi * 16;
      h8 blo = *(const h8*)(wRow);
      h8 bhi = *(const h8*)(wRow + 8);
      acc[s] = __builtin_amdgcn_wmma_f32_16x16x32_f16(
          false, a, false, cat8(blo, bhi), (short)0, acc[s], false, false);
    }
  }

  // LSTM cell, in-register: lane holds column j = j0+col, rows b = r + 8*hi
#pragma unroll
  for (int r = 0; r < 8; ++r) {
    int b = r + hi * 8;
    long ci = (long)b * LSTM_H + j0 + col;
    float si = sigmoidf_(acc[0][r]);
    float sf = sigmoidf_(acc[1][r]);
    float tg = tanhf(acc[2][r]);
    float so = sigmoidf_(acc[3][r]);
    float cn = sf * c[ci] + si * tg;
    c[ci] = cn;
    _Float16 hh = (_Float16)(so * tanhf(cn));
    hnext[ci] = hh;
    hs[(long)(t * LSTM_B + b) * LSTM_H + j0 + col] = hh;
  }
}

// ---------------------------------------------------------------------------
// Host orchestration
// ---------------------------------------------------------------------------
static inline int cdiv(int a, int b) { return (a + b - 1) / b; }

extern "C" void kernel_launch(void* const* d_in, const int* in_sizes, int n_in,
                              void* d_out, int out_size, void* d_ws,
                              size_t ws_size, hipStream_t stream) {
  (void)in_sizes; (void)n_in; (void)out_size; (void)ws_size;
  const int B = LSTM_B, T = LSTM_T, V = LSTM_V, E = LSTM_E, H = LSTM_H;
  const int G = 4 * H, M = B * T;

  const int*   tokens = (const int*)d_in[0];
  const float* emb    = (const float*)d_in[1];
  const float* W_ih0  = (const float*)d_in[2];
  const float* W_hh0  = (const float*)d_in[3];
  const float* b_ih0  = (const float*)d_in[4];
  const float* b_hh0  = (const float*)d_in[5];
  const float* W_ih1  = (const float*)d_in[6];
  const float* W_hh1  = (const float*)d_in[7];
  const float* b_ih1  = (const float*)d_in[8];
  const float* b_hh1  = (const float*)d_in[9];
  const float* W_out  = (const float*)d_in[10];
  const float* b_out  = (const float*)d_in[11];
  float* logits = (float*)d_out;

  // workspace bump allocator (256B aligned)
  char* ws = (char*)d_ws;
  size_t off = 0;
  auto take = [&](size_t bytes) -> char* {
    char* p = ws + off;
    off = (off + bytes + 255) & ~(size_t)255;
    return p;
  };
  _Float16* xe    = (_Float16*)take((size_t)M * E * 2);  // (T,B,E) f16
  _Float16* wih0h = (_Float16*)take((size_t)G * E * 2);
  _Float16* whh0h = (_Float16*)take((size_t)G * H * 2);
  _Float16* wih1h = (_Float16*)take((size_t)G * H * 2);
  _Float16* whh1h = (_Float16*)take((size_t)G * H * 2);
  _Float16* wouth = (_Float16*)take((size_t)V * H * 2);
  float*    gi    = (float*)take((size_t)M * G * 4);     // (T,B,4H), reused per layer
  _Float16* hs0   = (_Float16*)take((size_t)M * H * 2);  // (T,B,H) f16
  _Float16* hs1   = (_Float16*)take((size_t)M * H * 2);
  float*    cbuf  = (float*)take((size_t)B * H * 4);
  _Float16* hA    = (_Float16*)take((size_t)B * H * 2);  // h ping-pong
  _Float16* hB    = (_Float16*)take((size_t)B * H * 2);
  float*    bias0 = (float*)take((size_t)G * 4);
  float*    bias1 = (float*)take((size_t)G * 4);

  auto conv = [&](const float* s, _Float16* d, int n) {
    k_f32_to_f16<<<cdiv(n, 256), 256, 0, stream>>>(s, d, n);
  };
  auto gemm = [&](const _Float16* A, int lda, const _Float16* Wm, int ldw,
                  const float* bias, float* D, int ldd, int Mm, int Nn, int Kk,
                  int rB, int rT) {
    dim3 grid(Nn / BN, Mm / BM);
    wmma_gemm_tiled<<<grid, 256, 0, stream>>>(A, lda, Wm, ldw, bias, D, ldd,
                                              Kk, rB, rT);
  };

  // ---- prep: weight conversions, biases, embedding ----
  conv(W_ih0, wih0h, G * E);
  conv(W_hh0, whh0h, G * H);
  conv(W_ih1, wih1h, G * H);
  conv(W_hh1, whh1h, G * H);
  conv(W_out, wouth, V * H);
  k_bias_sum<<<cdiv(G, 256), 256, 0, stream>>>(b_ih0, b_hh0, bias0, G);
  k_bias_sum<<<cdiv(G, 256), 256, 0, stream>>>(b_ih1, b_hh1, bias1, G);
  k_embed<<<cdiv(T * B * E, 256), 256, 0, stream>>>(tokens, emb, xe);

  // ---- layer 0 ----
  gemm(xe, E, wih0h, E, bias0, gi, G, M, G, E, 0, 0);  // gi = x@W_ih0^T + b
  k_zero_state<<<cdiv(B * H, 256), 256, 0, stream>>>(cbuf, hA, hB);
  {
    const _Float16* hin = hA;
    _Float16* hout = hB;
    for (int t = 0; t < T; ++t) {
      k_lstm_step<<<8, 256, 0, stream>>>(hin, whh0h, gi + (size_t)t * B * G,
                                         cbuf, hout, hs0, t);
      const _Float16* tmp = hout; hout = (_Float16*)hin; hin = tmp;
    }
  }

  // ---- layer 1 ----
  gemm(hs0, H, wih1h, H, bias1, gi, G, M, G, H, 0, 0);
  k_zero_state<<<cdiv(B * H, 256), 256, 0, stream>>>(cbuf, hA, hB);
  {
    const _Float16* hin = hA;
    _Float16* hout = hB;
    for (int t = 0; t < T; ++t) {
      k_lstm_step<<<8, 256, 0, stream>>>(hin, whh1h, gi + (size_t)t * B * G,
                                         cbuf, hout, hs1, t);
      const _Float16* tmp = hout; hout = (_Float16*)hin; hin = tmp;
    }
  }

  // ---- output projection: logits(B,T,V) = hs1 @ W_out^T + b_out ----
  gemm(hs1, H, wouth, H, b_out, logits, V, M, V, H, B, T);
}
